// AllAttention_1013612282140
// MI455X (gfx1250) — matrix operand order repaired and checked by reading, not compiled, not run on hardware
//
#include <hip/hip_runtime.h>
#include <hip/hip_bf16.h>

// Problem sizes (fixed by the reference)
#define RNN_SIZE 1024
#define ATT_HID  512
#define BATCH    256
#define ATT_SIZE 196
#define NROWS    (BATCH * ATT_SIZE)   // 50176 flattened (b,a) rows
#define ROWS_PER_BLK 64               // 50176 / 64 = 784 blocks exactly
#define KCHUNK   32                   // K per WMMA step (f16 16x16x32)
#define KSTEPS   (RNN_SIZE / KCHUNK)  // 32

typedef __attribute__((ext_vector_type(16))) _Float16 v16h;
typedef __attribute__((ext_vector_type(2)))  __fp16   fp16x2;
typedef __attribute__((ext_vector_type(8)))  float    v8f;
typedef __attribute__((ext_vector_type(4)))  unsigned uint32x4;
typedef __attribute__((ext_vector_type(8)))  int      int32x8;
typedef __attribute__((ext_vector_type(4)))  int      int32x4;

// Fragment-ready LDS layout: per (row|col) a 16-dword record padded to 20 dwords
// (16-byte aligned records; each half-wave's b128 footprint tiles all 64 banks).
// Record position [hi*8 + j] holds the dword lane-half hi needs in A/B VGPR j.
#define FR_STRIDE 20
#define A_DWORDS  (ROWS_PER_BLK * FR_STRIDE)   // 1280 dwords / buffer
#define B_DWORDS  (ATT_HID * FR_STRIDE)        // 10240 dwords / buffer (8192 data + TDM pad)

#if defined(__has_builtin)
# if __has_builtin(__builtin_amdgcn_tensor_load_to_lds) && __has_builtin(__builtin_amdgcn_s_wait_tensorcnt)
#  define HAVE_TDM 1
# endif
#endif

__device__ __forceinline__ unsigned pack2h(float a, float b) {
    union { fp16x2 h; unsigned u; } p;
    p.h = __builtin_amdgcn_cvt_pkrtz(a, b);   // one v_cvt_pk_rtz_f16_f32
    return p.u;
}

// ---------------------------------------------------------------------------
// Kernel 1: pre-convert Wf (fp32 [1024,512]) -> f16 K-pair dwords, laid out
// per K-step in TDM-ready record order:
//   WfP[(kc*512 + n)*16 + kp] = {f16(Wf[2K][n]), f16(Wf[2K+1][n])}, K = kc*16+kp
// Each K-step's 32KB B tile is then one contiguous block -> one TDM descriptor.
// ---------------------------------------------------------------------------
__global__ void pack_wf(const float* __restrict__ Wf, unsigned* __restrict__ WfP) {
    int i = blockIdx.x * 256 + threadIdx.x;   // 262144 items
    int n  = i & 511;                          // fastest -> coalesced Wf reads
    int kp = (i >> 9) & 15;
    int kc = i >> 13;
    int k2 = kc * 16 + kp;
    float lo = Wf[(2 * k2) * ATT_HID + n];
    float hi = Wf[(2 * k2 + 1) * ATT_HID + n];
    WfP[(size_t)((kc << 9) + n) * 16 + kp] = pack2h(lo, hi);
}

// ---------------------------------------------------------------------------
// Kernel 2: attH[b,n] = bh[n] + sum_r h[b,r] * Wh[r,n]   (small, VALU fp32)
// ---------------------------------------------------------------------------
__global__ void atth(const float* __restrict__ h, const float* __restrict__ Wh,
                     const float* __restrict__ bh, float* __restrict__ attH) {
    int n = blockIdx.x * 256 + threadIdx.x;
    int b = blockIdx.y;
    const float* hb = h + (size_t)b * RNN_SIZE;
    float acc = bh[n];
    #pragma unroll 4
    for (int r = 0; r < RNN_SIZE; ++r)
        acc = fmaf(hb[r], Wh[(size_t)r * ATT_HID + n], acc);
    attH[(size_t)b * ATT_HID + n] = acc;
}

// ---------------------------------------------------------------------------
// Kernel 3: fused scores (the heavy GEMM).
//   scores[row] = ba + sum_n wa[n] * tanh( (feats_row . Wf[:,n]) + bf[n] + attH[b,n] )
// 512 threads = 16 waves; tile 64 rows x 512 cols, K=1024 in 32 steps.
// Wave w: mSub = w&3 (16 rows), nq = w>>2 (128 cols = 8 WMMA n-tiles).
// B tile staged by ONE Tensor-Data-Mover descriptor per step (pad_enable
// reproduces the 20-dword bank padding); A staged as a split-phase pipeline:
// global_load_b128 issued before the WMMA burst, cvt+ds_store after it, so
// both the TDM DMA and the A global load latency hide under the 8 WMMAs.
// ---------------------------------------------------------------------------
__global__ void scores_wmma(const float* __restrict__ feats,
                            const unsigned* __restrict__ WfP,
                            const float* __restrict__ bf,
                            const float* __restrict__ attH,
                            const float* __restrict__ wa,
                            const float* __restrict__ ba,
                            float* __restrict__ scores) {
    __shared__ unsigned lA[2 * A_DWORDS];
    __shared__ unsigned lB[2 * B_DWORDS];
    __shared__ float    sc[ROWS_PER_BLK];

    const int tid   = threadIdx.x;
    const int lane  = tid & 31;
    const int wave  = tid >> 5;
    const int l15   = lane & 15;
    const int hi    = lane >> 4;
    const int mSub  = wave & 3;
    const int nq    = wave >> 2;
    const int rowBase = blockIdx.x * ROWS_PER_BLK;

    if (tid < ROWS_PER_BLK) sc[tid] = 0.0f;

    v8f acc[8];
    #pragma unroll
    for (int nt = 0; nt < 8; ++nt) acc[nt] = (v8f){0,0,0,0,0,0,0,0};

    const unsigned ldsB0 = (unsigned)(size_t)&lB[0];   // LDS byte offset of lB

    // ---- B staging: one TDM descriptor per K-step (wave 0 issues) ----
    auto issueB = [&](int kc, int buf) {
#if HAVE_TDM
        if (wave == 0) {
            unsigned long long ga =
                (unsigned long long)(size_t)(WfP + (size_t)kc * 8192);
            uint32x4 g0;
            g0.x = 1u;                                   // count=1 valid user D#
            g0.y = ldsB0 + (unsigned)(buf * B_DWORDS) * 4u;   // lds_addr
            g0.z = (unsigned)ga;                         // global_addr[31:0]
            g0.w = (unsigned)(ga >> 32) | (2u << 30);    // global_addr[56:32]|type=2
            int32x8 g1;
            // data_size=4B(2), pad_enable, pad_interval=16dw(3), pad_amount=4dw(3)
            g1[0] = (int)((2u << 16) | (1u << 20) | (3u << 22) | (3u << 25));
            g1[1] = (int)(8192u << 16);                  // tensor_dim0 = 8192 dwords
            g1[2] = 0;
            g1[3] = (int)(8192u << 16);                  // tile_dim0 = 8192 dwords
            g1[4] = 0; g1[5] = 0; g1[6] = 0; g1[7] = 0;
            int32x4 z4 = {0, 0, 0, 0};
#if __clang_major__ >= 23
            int32x8 z8 = {0, 0, 0, 0, 0, 0, 0, 0};
            __builtin_amdgcn_tensor_load_to_lds(g0, g1, z4, z4, z8, 0);
#else
            __builtin_amdgcn_tensor_load_to_lds(g0, g1, z4, z4, 0);
#endif
        }
#else
        // fallback: per-lane async global->LDS dword copies
        const unsigned* wsrc = WfP + (size_t)kc * 8192 + tid * 16;
        unsigned dbase = ldsB0 + (unsigned)(buf * B_DWORDS + tid * FR_STRIDE) * 4u;
        #pragma unroll
        for (int kp = 0; kp < 16; ++kp) {
            unsigned dst = dbase + kp * 4u;
            const unsigned* src = wsrc + kp;
            asm volatile("global_load_async_to_lds_b32 %0, %1, off"
                         :: "v"(dst), "v"(src) : "memory");
        }
#endif
    };
    auto waitB = [&]() {
#if HAVE_TDM
        if (wave == 0) __builtin_amdgcn_s_wait_tensorcnt(0);
#else
        asm volatile("s_wait_asynccnt 0x0" ::: "memory");
#endif
    };

    // ---- A staging, split in two phases so the load hides under WMMAs ----
    const int aRow = tid >> 3;           // 64 rows
    const int aDp  = tid & 7;            // dword-pair within record (kp = 2*dp)
    const int aGrp  = aDp >> 1;                                    // kp>>2
    const int aPgrp = aGrp;                                        // identity: contiguous k
    const int aPos  = aPgrp * 4 + ((2 * aDp) & 3);
    auto loadAreg = [&](int kc) -> float4 {
        const float* src = feats + (size_t)(rowBase + aRow) * RNN_SIZE
                                 + kc * KCHUNK + aDp * 4;
        return *(const float4*)src;
    };
    auto storeA = [&](float4 f, int buf) {
        uint2 v2 = make_uint2(pack2h(f.x, f.y), pack2h(f.z, f.w));
        *(uint2*)&lA[buf * A_DWORDS + aRow * FR_STRIDE + aPos] = v2;
    };

    // prologue
    issueB(0, 0);
    storeA(loadAreg(0), 0);
    waitB();
    __syncthreads();

    union Frag { unsigned u[8]; uint4 q[2]; v16h v; };
    const int arowOff = (mSub * 16 + l15) * FR_STRIDE + hi * 8;

    auto loadB = [&](Frag& f, int cur, int nt) {
        int n = nq * 128 + nt * 16 + l15;
        const uint4* bp = (const uint4*)&lB[cur * B_DWORDS + n * FR_STRIDE + hi * 8];
        f.q[0] = bp[0];
        f.q[1] = bp[1];
    };

    for (int kc = 0; kc < KSTEPS; ++kc) {
        const int cur = kc & 1, nxt = cur ^ 1;
        const bool pf = (kc + 1 < KSTEPS);

        float4 fA;
        if (pf) {
            issueB(kc + 1, nxt);        // TDM DMA runs under the WMMA burst
            fA = loadAreg(kc + 1);      // global load in flight under WMMAs
        }

        Frag af;
        const uint4* ap = (const uint4*)&lA[cur * A_DWORDS + arowOff];
        af.q[0] = ap[0];
        af.q[1] = ap[1];

        Frag b0, b1;                    // 2-deep register double buffer
        loadB(b0, cur, 0);
        #pragma unroll
        for (int nt = 0; nt < 8; ++nt) {
            Frag& c = (nt & 1) ? b1 : b0;
            Frag& n2 = (nt & 1) ? b0 : b1;
            if (nt < 7) loadB(n2, cur, nt + 1);
            acc[nt] = __builtin_amdgcn_wmma_f32_16x16x32_f16(
                false, af.v, false, c.v, (short)0, acc[nt], false, false);
        }

        if (pf) storeA(fA, nxt);        // wait loadcnt + cvt + ds_store here
        waitB();
        __syncthreads();
    }

    // ---- epilogue: +bf +attH, tanh, *wa, reduce over n ----
    // C layout: lane(0-15) VGPR v -> M=v ; lane(16-31) VGPR v -> M=8+v ; N=l15
    #pragma unroll
    for (int v = 0; v < 8; ++v) {
        int mloc = mSub * 16 + 8 * hi + v;
        int row  = rowBase + mloc;
        int b    = row / ATT_SIZE;
        float partial = 0.0f;
        #pragma unroll
        for (int nt = 0; nt < 8; ++nt) {
            int n = nq * 128 + nt * 16 + l15;
            float x = acc[nt][v] + bf[n] + attH[(size_t)b * ATT_HID + n];
            partial += wa[n] * tanhf(x);
        }
        #pragma unroll
        for (int m = 1; m < 16; m <<= 1)
            partial += __shfl_xor(partial, m, 32);
        if (l15 == 0) atomicAdd(&sc[mloc], partial);   // 4 nq-waves accumulate
    }
    __syncthreads();
    if (tid < ROWS_PER_BLK)
        scores[rowBase + tid] = sc[tid] + ba[0];
}

// ---------------------------------------------------------------------------
// Kernel 4: per-batch softmax over 196 scores + weighted sum over feats.
// 256 blocks (one per b), 256 threads; float4-coalesced, HBM-streaming.
// ---------------------------------------------------------------------------
__global__ void softmax_wsum(const float* __restrict__ feats,
                             const float* __restrict__ scores,
                             float* __restrict__ out) {
    __shared__ float w[256];
    __shared__ float red[256];
    int b = blockIdx.x, t = threadIdx.x;

    float s = (t < ATT_SIZE) ? scores[b * ATT_SIZE + t] : -3.402823e38f;
    red[t] = s; __syncthreads();
    for (int off = 128; off > 0; off >>= 1) {
        if (t < off) red[t] = fmaxf(red[t], red[t + off]);
        __syncthreads();
    }
    float mx = red[0]; __syncthreads();

    float e = (t < ATT_SIZE) ? __expf(s - mx) : 0.0f;
    red[t] = e; __syncthreads();
    for (int off = 128; off > 0; off >>= 1) {
        if (t < off) red[t] += red[t + off];
        __syncthreads();
    }
    w[t] = e * (1.0f / red[0]);
    __syncthreads();

    const float4* fb = (const float4*)(feats + (size_t)b * ATT_SIZE * RNN_SIZE) + t;
    float4 acc = make_float4(0.f, 0.f, 0.f, 0.f);
    for (int a = 0; a < ATT_SIZE; ++a) {
        if (a + 4 < ATT_SIZE)
            __builtin_prefetch(&fb[(size_t)(a + 4) * (RNN_SIZE / 4)], 0, 0);
        float4 f  = fb[(size_t)a * (RNN_SIZE / 4)];
        float  wv = w[a];
        acc.x = fmaf(wv, f.x, acc.x);
        acc.y = fmaf(wv, f.y, acc.y);
        acc.z = fmaf(wv, f.z, acc.z);
        acc.w = fmaf(wv, f.w, acc.w);
    }
    ((float4*)out)[(size_t)b * (RNN_SIZE / 4) + t] = acc;
}

// ---------------------------------------------------------------------------
extern "C" void kernel_launch(void* const* d_in, const int* in_sizes, int n_in,
                              void* d_out, int out_size, void* d_ws, size_t ws_size,
                              hipStream_t stream) {
    const float* h     = (const float*)d_in[0];
    const float* feats = (const float*)d_in[1];
    const float* Wf    = (const float*)d_in[2];
    const float* bf    = (const float*)d_in[3];
    const float* Wh    = (const float*)d_in[4];
    const float* bh    = (const float*)d_in[5];
    const float* wa    = (const float*)d_in[6];
    const float* ba    = (const float*)d_in[7];
    float* out = (float*)d_out;

    char* ws = (char*)d_ws;
    unsigned* WfP  = (unsigned*)ws;                                  // 1 MB
    float*    attH = (float*)(ws + (1u << 20));                      // 512 KB
    float*    scr  = (float*)(ws + (1u << 20) + (512u << 10));       // ~200 KB

    pack_wf<<<(512 * 512) / 256, 256, 0, stream>>>(Wf, WfP);
    atth<<<dim3(ATT_HID / 256, BATCH), 256, 0, stream>>>(h, Wh, bh, attH);
    scores_wmma<<<NROWS / ROWS_PER_BLK, 512, 0, stream>>>(feats, WfP, bf, attH, wa, ba, scr);
    softmax_wsum<<<BATCH, 256, 0, stream>>>(feats, scr, out);
}
